// PhaseDecoder_14121852469305
// MI455X (gfx1250) — compile-verified, matrix operand order checked by reading, not run on hardware
//
#include <hip/hip_runtime.h>
#include <hip/hip_bf16.h>
#include <math.h>

// ---------------------------------------------------------------------------
// PhaseDecoder on MI455X (gfx1250, wave32, WMMA).
// All GEMMs use v_wmma_f32_16x16x32_bf16 (fp32 accumulate).
// Dominant cost: logits GEMM [4096,512]x[512,32000] = 134 GFLOP + 524 MB
// fp32 output write (~22us floor at 23.3 TB/s) -> bf16 tensor path chosen.
// Round-2 changes: spill-free loop nests (kk-outer / acc-tile-array inner),
// hoisted LDS A-fragments in attention phase C, async-LDS A-strip staging in
// the FF GEMM (all waves of a block share one strip).
// ---------------------------------------------------------------------------

#define B_  32
#define S_  128
#define D_  256
#define K2_ 512           // 2*D (real|imag concatenated K dimension)
#define NT_ 4096          // B*S tokens
#define V_  32000
#define EPS_ 1e-5f

typedef __attribute__((ext_vector_type(16))) __bf16 v16bf;
typedef __attribute__((ext_vector_type(8)))  float  v8f;

// ---- 16x32 bf16 fragment loader (A-layout; B uses same layout on [N,K] data)
// ISA 7.12.2: lanes 0-15 hold M=0..15, lanes 16-31 same M with K offset +8;
// VGPR v holds K pair: k = (v>=4?16:0) + (v%4)*2 + pairbit (+8 for hi half).
template <typename T>
__device__ inline v16bf frag16(const T* __restrict__ base, int ld, int row0, int k0) {
  const int lane = threadIdx.x & 31;
  const int m    = lane & 15;
  const int hf   = lane >> 4;
  v16bf f;
#pragma unroll
  for (int e = 0; e < 16; ++e) {
    const int vv = e >> 1;
    const int k  = k0 + (hf << 3) + ((vv >= 4) ? 16 : 0) + ((vv & 3) << 1) + (e & 1);
    f[e] = (__bf16)base[(size_t)(row0 + m) * (size_t)ld + k];
  }
  return f;
}

__device__ inline v8f wmma_bf16(v16bf a, v16bf b, v8f c) {
  return __builtin_amdgcn_wmma_f32_16x16x32_bf16(false, a, false, b, (short)0, c,
                                                 false, false);
}

// Stage `bytes` (multiple of 256*16) from global to LDS with async-LDS loads.
__device__ inline void async_stage(unsigned lbase, const char* src, int bytes) {
  const int tid = threadIdx.x;                     // blockDim.x == 256
  const int chunks = bytes >> 4;                   // 16B units
#pragma unroll
  for (int j = tid; j < chunks; j += 256) {
    unsigned    lofs = lbase + (unsigned)(j * 16);
    const char* g    = src + j * 16;
    asm volatile("global_load_async_to_lds_b128 %0, %1, off"
                 :: "v"(lofs), "v"(g) : "memory");
  }
  asm volatile("s_wait_asynccnt 0" ::: "memory");
}

// ---------------------------------------------------------------------------
// Kernel 1: pack complex FF matrix into real [512,512] block matrix, stored
// transposed [N,K] bf16:  out_r = sr@fr - si@fi ; out_i = sr@fi + si@fr
__global__ void prep_ff_kernel(const float* __restrict__ fr,
                               const float* __restrict__ fi,
                               __bf16* __restrict__ F2T) {
  int idx = blockIdx.x * 256 + threadIdx.x;          // 512*512
  int n = idx >> 9, k = idx & 511;
  float v;
  if (n < D_) {                                       // real output column n
    v = (k < D_) ? fr[k * D_ + n] : -fi[(k - D_) * D_ + n];
  } else {                                            // imag output column n-256
    int nn = n - D_;
    v = (k < D_) ? fi[k * D_ + nn] : fr[(k - D_) * D_ + nn];
  }
  F2T[(size_t)n * K2_ + k] = (__bf16)v;
}

// Kernel 2: Wcat[v, k] = k<256 ? wr[v,k] : wi[v,k-256]   (bf16, [N,K] layout)
__global__ void prep_w_kernel(const float* __restrict__ wr,
                              const float* __restrict__ wi,
                              __bf16* __restrict__ Wcat) {
  size_t idx = (size_t)blockIdx.x * 256 + threadIdx.x; // 32000*512
  int vv = (int)(idx >> 9), k = (int)(idx & 511);
  float x = (k < D_) ? wr[(size_t)vv * D_ + k] : wi[(size_t)vv * D_ + (k - D_)];
  Wcat[idx] = (__bf16)x;
}

// ---------------------------------------------------------------------------
// Kernel 3: embedding gather + tanh + positional phase + q/k/v rotations.
// Outputs: state2 f32 [NT,512], Q2/K2 bf16 [NT,512], vrT/viT bf16 [B,D,S].
__global__ void embed_kernel(const int* __restrict__ x,
                             const float* __restrict__ emb,
                             const float* __restrict__ qrot,
                             const float* __restrict__ krot,
                             const float* __restrict__ vrot,
                             float* __restrict__ state2,
                             __bf16* __restrict__ Q2,
                             __bf16* __restrict__ K2,
                             __bf16* __restrict__ vrT,
                             __bf16* __restrict__ viT) {
  const int tok = blockIdx.x;            // 0..4095
  const int d   = threadIdx.x;           // 0..255
  const int b   = tok >> 7;
  const int s   = tok & 127;

  const int id  = x[tok];
  const float mag = tanhf(emb[(size_t)id * D_ + d]);

  const float freq  = powf(10000.0f, -(float)d / (float)D_);
  const float phase = (float)s * freq * 3.14159265358979323846f;
  float cp, sp;  __sincosf(phase, &sp, &cp);
  const float sr = mag * cp;
  const float si = mag * sp;

  state2[(size_t)tok * K2_ + d]      = sr;
  state2[(size_t)tok * K2_ + D_ + d] = si;

  float c, sn;
  __sincosf(qrot[d], &sn, &c);
  Q2[(size_t)tok * K2_ + d]      = (__bf16)(sr * c - si * sn);
  Q2[(size_t)tok * K2_ + D_ + d] = (__bf16)(sr * sn + si * c);
  __sincosf(krot[d], &sn, &c);
  K2[(size_t)tok * K2_ + d]      = (__bf16)(sr * c - si * sn);
  K2[(size_t)tok * K2_ + D_ + d] = (__bf16)(sr * sn + si * c);
  __sincosf(vrot[d], &sn, &c);
  vrT[((size_t)b * D_ + d) * S_ + s] = (__bf16)(sr * c - si * sn);
  viT[((size_t)b * D_ + d) * S_ + s] = (__bf16)(sr * sn + si * c);
}

// ---------------------------------------------------------------------------
// Kernel 4: attention. 1 block per batch, 8 waves; wave w owns rows 16w..16w+15.
//  A) logits strip via WMMA (kk-outer, acc[8] tile array -> no spills),
//     scale 0.5 (=8/sqrt(D)), causal mask -> LDS
//  B) per-row softmax in LDS
//  C) attn_w @ V (real & imag) via WMMA (K=128), A-frags hoisted, + residual
__global__ void attn_kernel(const float* __restrict__ state2,
                            const __bf16* __restrict__ Q2,
                            const __bf16* __restrict__ K2,
                            const __bf16* __restrict__ vrT,
                            const __bf16* __restrict__ viT,
                            float* __restrict__ attn2) {
  __shared__ float sWt[8][16][S_];                 // 64 KB
  const int b    = blockIdx.x;
  const int wave = threadIdx.x >> 5;
  const int lane = threadIdx.x & 31;
  const int hf   = lane >> 4;
  const int nn   = lane & 15;

  const __bf16* Qb = Q2 + (size_t)b * S_ * K2_;
  const __bf16* Kb = K2 + (size_t)b * S_ * K2_;

  // --- Phase A: 16x128 logits strip; one A frag live, 8 accumulators
  v8f acc[8];
#pragma unroll
  for (int t = 0; t < 8; ++t) acc[t] = (v8f){};
  for (int kk = 0; kk < 16; ++kk) {
    const v16bf a = frag16(Qb, K2_, 16 * wave, 32 * kk);
#pragma unroll
    for (int t = 0; t < 8; ++t) {
      v16bf bb = frag16(Kb, K2_, 16 * t, 32 * kk);
      acc[t] = wmma_bf16(a, bb, acc[t]);
    }
  }
#pragma unroll
  for (int t = 0; t < 8; ++t) {
#pragma unroll
    for (int e = 0; e < 8; ++e) {
      const int m = e + 8 * hf;
      const int r = 16 * wave + m;
      const int c = 16 * t + nn;
      float v = acc[t][e] * 0.5f;                  // (1/sqrt(256)) * 8
      if (c > r) v = -3.0e38f;                     // causal mask
      sWt[wave][m][c] = v;
    }
  }
  __syncthreads();

  // --- Phase B: row softmax (lane < 16 handles one row)
  if (lane < 16) {
    float mx = -3.0e38f;
    for (int c = 0; c < S_; ++c) mx = fmaxf(mx, sWt[wave][lane][c]);
    float sum = 0.f;
    for (int c = 0; c < S_; ++c) sum += __expf(sWt[wave][lane][c] - mx);
    const float inv = 1.0f / sum;
    for (int c = 0; c < S_; ++c)
      sWt[wave][lane][c] = __expf(sWt[wave][lane][c] - mx) * inv;
  }
  __syncthreads();

  // --- Phase C: attn_w @ V (+ residual); A fragments hoisted (4x8 VGPRs)
  v16bf aw[4];
#pragma unroll
  for (int kk = 0; kk < 4; ++kk)
    aw[kk] = frag16(&sWt[wave][0][0], S_, 0, 32 * kk);   // LDS, f32->bf16

  for (int hr = 0; hr < 2; ++hr) {                 // 0: real, 1: imag
    const __bf16* Vb = (hr ? viT : vrT) + (size_t)b * D_ * S_;
    for (int nt = 0; nt < 16; ++nt) {              // D/16 output column tiles
      v8f oc = {};
#pragma unroll
      for (int kk = 0; kk < 4; ++kk) {
        v16bf bb = frag16(Vb, S_, 16 * nt, 32 * kk);
        oc = wmma_bf16(aw[kk], bb, oc);
      }
#pragma unroll
      for (int e = 0; e < 8; ++e) {
        const int m   = e + 8 * hf;
        const int tok = b * S_ + 16 * wave + m;
        const int col = hr * D_ + 16 * nt + nn;
        attn2[(size_t)tok * K2_ + col] =
            state2[(size_t)tok * K2_ + col] + oc[e];
      }
    }
  }
}

// ---------------------------------------------------------------------------
// Kernel 5: FF GEMM [4096,512] x [512,512] (B stored [N,K]) -> ff2 f32.
// A strip (16x512 f32 = 32 KB) async-staged to LDS, shared by all 8 waves.
__global__ void ff_kernel(const float* __restrict__ attn2,
                          const __bf16* __restrict__ F2T,
                          float* __restrict__ ff2) {
  __shared__ float sAf[16 * K2_];                   // 32 KB
  const int mtile = blockIdx.x;                     // 0..255
  const int wave  = threadIdx.x >> 5;
  const int lane  = threadIdx.x & 31;
  const int ntile = blockIdx.y * 8 + wave;          // 0..31

  async_stage((unsigned)(size_t)(&sAf[0]),
              (const char*)(attn2 + (size_t)mtile * 16 * K2_), 16 * K2_ * 4);
  __syncthreads();

  v8f acc = {};
#pragma unroll
  for (int kk = 0; kk < 16; ++kk) {
    v16bf a  = frag16(sAf, K2_, 0,          32 * kk);    // LDS f32 -> bf16
    v16bf bb = frag16(F2T, K2_, 16 * ntile, 32 * kk);
    acc = wmma_bf16(a, bb, acc);
  }
  const int hf = lane >> 4, nn = lane & 15;
#pragma unroll
  for (int e = 0; e < 8; ++e) {
    const int m = e + 8 * hf;
    ff2[(size_t)(16 * mtile + m) * K2_ + 16 * ntile + nn] = acc[e];
  }
}

// ---------------------------------------------------------------------------
// Kernel 6: complex_norm (per token: mean/std(ddof=1) of |z|, tanh scale) -> bf16
__global__ void norm_kernel(const float* __restrict__ ff2,
                            __bf16* __restrict__ S2) {
  __shared__ float rs[256];
  __shared__ float rq[256];
  const int tok = blockIdx.x;
  const int d   = threadIdx.x;
  const float r = ff2[(size_t)tok * K2_ + d];
  const float i = ff2[(size_t)tok * K2_ + D_ + d];
  const float mag = sqrtf(r * r + i * i);
  rs[d] = mag; rq[d] = mag * mag;
  __syncthreads();
  for (int st = 128; st > 0; st >>= 1) {
    if (d < st) { rs[d] += rs[d + st]; rq[d] += rq[d + st]; }
    __syncthreads();
  }
  const float mean = rs[0] * (1.0f / D_);
  float var = (rq[0] - (float)D_ * mean * mean) * (1.0f / (D_ - 1));
  var = fmaxf(var, 0.0f);
  const float nrm   = (mag - mean) / (sqrtf(var) + EPS_);
  const float scale = tanhf(nrm) / (mag + EPS_);
  S2[(size_t)tok * K2_ + d]      = (__bf16)(r * scale);
  S2[(size_t)tok * K2_ + D_ + d] = (__bf16)(i * scale);
}

// ---------------------------------------------------------------------------
// Kernel 7: logits GEMM [4096,512] x [512,32000] + biases.
// A strip (16x512 bf16 = 16KB) staged to LDS with async-LDS loads (CDNA5 path).
__global__ void logits_kernel(const __bf16* __restrict__ S2,
                              const __bf16* __restrict__ Wcat,
                              const float* __restrict__ br,
                              const float* __restrict__ bi,
                              float* __restrict__ out) {
  __shared__ __bf16 sA[16 * K2_];                   // 16 KB
  const int mtile = blockIdx.x;                     // 0..255
  const int wave  = threadIdx.x >> 5;
  const int lane  = threadIdx.x & 31;

  async_stage((unsigned)(size_t)(&sA[0]),
              (const char*)(S2 + (size_t)mtile * 16 * K2_), 16 * K2_ * 2);
  __syncthreads();

  const int ntile = blockIdx.y * 8 + wave;          // 0..1999
  v8f acc = {};
#pragma unroll
  for (int kk = 0; kk < 16; ++kk) {
    __builtin_prefetch(&Wcat[(size_t)(16 * ntile + (lane & 15)) * K2_ +
                             32 * ((kk + 2) & 15)], 0, 1);
    v16bf a  = frag16(sA,   K2_, 0,          32 * kk);   // LDS (ds_load)
    v16bf bb = frag16(Wcat, K2_, 16 * ntile, 32 * kk);
    acc = wmma_bf16(a, bb, acc);
  }
  const int hf = lane >> 4, nn = lane & 15;
#pragma unroll
  for (int e = 0; e < 8; ++e) {
    const int m   = e + 8 * hf;
    const int col = 16 * ntile + nn;
    out[(size_t)(16 * mtile + m) * V_ + col] = acc[e] + br[col] + bi[col];
  }
}

// ---------------------------------------------------------------------------
extern "C" void kernel_launch(void* const* d_in, const int* in_sizes, int n_in,
                              void* d_out, int out_size, void* d_ws, size_t ws_size,
                              hipStream_t stream) {
  const int*   x     = (const int*)  d_in[0];
  const float* emb   = (const float*)d_in[1];
  const float* qrot  = (const float*)d_in[2];
  const float* krot  = (const float*)d_in[3];
  const float* vrot  = (const float*)d_in[4];
  const float* ffr   = (const float*)d_in[5];
  const float* ffi   = (const float*)d_in[6];
  const float* wr    = (const float*)d_in[7];
  const float* br    = (const float*)d_in[8];
  const float* wi    = (const float*)d_in[9];
  const float* bi    = (const float*)d_in[10];
  float* out = (float*)d_out;

  char* w = (char*)d_ws;
  float*  state2 = (float*) (w + 0);           //  8 MB  [NT,512] f32
  __bf16* Q2     = (__bf16*)(w + 8388608);     //  4 MB  [NT,512] bf16
  __bf16* K2m    = (__bf16*)(w + 12582912);    //  4 MB
  __bf16* vrT    = (__bf16*)(w + 16777216);    //  2 MB  [B,D,S]
  __bf16* viT    = (__bf16*)(w + 18874368);    //  2 MB
  float*  attn2  = (float*) (w + 20971520);    //  8 MB  [NT,512] f32
  __bf16* F2T    = (__bf16*)(w + 29360128);    // 0.5 MB [512,512]
  float*  ff2    = (float*) (w + 29884416);    //  8 MB  [NT,512] f32
  __bf16* S2     = (__bf16*)(w + 38273024);    //  4 MB  [NT,512] bf16
  __bf16* Wcat   = (__bf16*)(w + 42467328);    // 32.8MB [V,512]  bf16

  prep_ff_kernel<<<dim3(K2_ * K2_ / 256), dim3(256), 0, stream>>>(ffr, ffi, F2T);
  prep_w_kernel <<<dim3(V_ * K2_ / 256),  dim3(256), 0, stream>>>(wr, wi, Wcat);
  embed_kernel  <<<dim3(NT_), dim3(D_), 0, stream>>>(x, emb, qrot, krot, vrot,
                                                     state2, Q2, K2m, vrT, viT);
  attn_kernel   <<<dim3(B_), dim3(256), 0, stream>>>(state2, Q2, K2m, vrT, viT, attn2);
  ff_kernel     <<<dim3(NT_ / 16, 4), dim3(256), 0, stream>>>(attn2, F2T, ff2);
  norm_kernel   <<<dim3(NT_), dim3(D_), 0, stream>>>(ff2, S2);
  logits_kernel <<<dim3(NT_ / 16, V_ / 16 / 8), dim3(256), 0, stream>>>(S2, Wcat,
                                                                        br, bi, out);
}
